// SparseAttentionAggregator_82325933130382
// MI455X (gfx1250) — matrix operand order repaired and checked by reading, not compiled, not run on hardware
//
#include <hip/hip_runtime.h>

typedef __attribute__((ext_vector_type(16))) _Float16 v16h;
typedef __attribute__((ext_vector_type(8)))  _Float16 v8h;
typedef __attribute__((ext_vector_type(2)))  _Float16 v2h;
typedef __attribute__((ext_vector_type(2)))  __fp16   v2r;   // cvt_pkrtz return type
typedef __attribute__((ext_vector_type(8)))  float    v8f;

#define B_    4
#define H_    16
#define S_    2048
#define D_    64
#define KB    64              // keys per tile
#define QW    16              // q rows per wave
#define WAVES 4               // waves per workgroup
#define QB    (QW * WAVES)    // q rows per workgroup

// scores are computed pre-scaled into the log2 domain:
//   s2 = (q.k) * (1/sqrt(D)) * log2(e);  p = exp2(s2 - m2)  ==  exp(s - m)
#define QSCALE  (0.125f * 1.44269504088896340736f)
// masked fill: exactly representable in f16 AND f32; with any live key masked p
// underflows to 0 (== reference exp(-1e9-m)); fully-masked row -> uniform p=1.
#define NEGM2   (-61440.0f)

union AFrag  { v16h v; v8h h[2]; };
union PkRow  { v2r  p[8]; v8h h[2]; };
union HalfU  { v2r  raw; v2h v; int i; };

__global__ __launch_bounds__(128)
void flash_attn_f16wmma_kernel(const float* __restrict__ Q,
                               const float* __restrict__ K,
                               const float* __restrict__ V,
                               const int*   __restrict__ M,
                               float*       __restrict__ O)
{
  __shared__ __align__(32) _Float16 sK [KB * D_];         // [key][d]   row-major
  __shared__ __align__(32) _Float16 sVt[D_ * KB];         // [d][key]   transposed
  __shared__ __align__(32) _Float16 sP [WAVES][QW * KB];  // per-wave P tile [row][key]

  const int tid  = threadIdx.x;
  const int wave = tid >> 5;
  const int lane = tid & 31;
  const int col  = lane & 15;   // column within 16 (N for C-layout, M for A-layout)
  const int half = lane >> 4;   // 0: lanes 0-15, 1: lanes 16-31

  const int bh    = blockIdx.y;       // b*H + h
  const int b     = bh / H_;
  const int qBase = blockIdx.x * QB + wave * QW;
  const size_t headOff = (size_t)bh * S_ * D_;

  // ---- Q A-operand fragments, pre-scaled by 1/sqrt(D)*log2(e) ----
  // A operand k order used here (16-bit, 16x32): lane m = col; halves 0-7 -> K = 16*half + i,
  //                           halves 8-15 -> K = 16*half + 8 + i
  AFrag aQ[2];
  {
    const float* qrow = Q + headOff + (size_t)(qBase + col) * D_;
    const int kb8 = half * 16;
#pragma unroll
    for (int c = 0; c < 2; ++c) {
#pragma unroll
      for (int i = 0; i < 8; ++i) {
        aQ[c].v[i]     = (_Float16)(qrow[c * 32 + kb8 + i]      * QSCALE);
        aQ[c].v[i + 8] = (_Float16)(qrow[c * 32 + kb8 + 8 + i] * QSCALE);
      }
    }
  }

  // all-ones B operand: one WMMA against it produces row-sums of P in every lane
  v16h bones;
#pragma unroll
  for (int i = 0; i < 16; ++i) bones[i] = (_Float16)1.0f;

  // ---- flash state ----
  const v8f vzero = {};
  v8f   acco[4] = { vzero, vzero, vzero, vzero };  // O accum: 4 chunks of 16 d-cols
  float mrow[8], lrow[8];
#pragma unroll
  for (int r = 0; r < 8; ++r) { mrow[r] = -INFINITY; lrow[r] = 0.0f; }

  const int kp     = tid >> 2;          // fill: key pair (2*kp, 2*kp+1)
  const int dstart = (tid & 3) * 16;    // fill: 16 consecutive d per thread

  for (int kb = 0; kb < S_ / KB; ++kb) {
    const int kBase = kb * KB;

    // ---- cooperative fill: f32 -> f16 (packed cvt) into LDS ----
    {
      const float* g0 = K + headOff + (size_t)(kBase + 2 * kp) * D_ + dstart;
      const float* g1 = g0 + D_;
      const float* h0 = V + headOff + (size_t)(kBase + 2 * kp) * D_ + dstart;
      const float* h1 = h0 + D_;
      float k0[16], k1[16], u0[16], u1[16];
#pragma unroll
      for (int i = 0; i < 4; ++i) {
        float4 a = ((const float4*)g0)[i], bb = ((const float4*)g1)[i];
        float4 cc = ((const float4*)h0)[i], d = ((const float4*)h1)[i];
        k0[4*i] = a.x;  k0[4*i+1] = a.y;  k0[4*i+2] = a.z;  k0[4*i+3] = a.w;
        k1[4*i] = bb.x; k1[4*i+1] = bb.y; k1[4*i+2] = bb.z; k1[4*i+3] = bb.w;
        u0[4*i] = cc.x; u0[4*i+1] = cc.y; u0[4*i+2] = cc.z; u0[4*i+3] = cc.w;
        u1[4*i] = d.x;  u1[4*i+1] = d.y;  u1[4*i+2] = d.z;  u1[4*i+3] = d.w;
      }
      // K rows: packed convert, two 16B stores per row
      PkRow r0, r1;
#pragma unroll
      for (int i = 0; i < 8; ++i) {
        r0.p[i] = __builtin_amdgcn_cvt_pkrtz(k0[2*i], k0[2*i+1]);
        r1.p[i] = __builtin_amdgcn_cvt_pkrtz(k1[2*i], k1[2*i+1]);
      }
      *(v8h*)&sK[(2*kp)     * D_ + dstart]     = r0.h[0];
      *(v8h*)&sK[(2*kp)     * D_ + dstart + 8] = r0.h[1];
      *(v8h*)&sK[(2*kp + 1) * D_ + dstart]     = r1.h[0];
      *(v8h*)&sK[(2*kp + 1) * D_ + dstart + 8] = r1.h[1];
      // V transposed: pack the two keys' values for each d -> one b32 store
#pragma unroll
      for (int i = 0; i < 16; ++i) {
        v2r pv = __builtin_amdgcn_cvt_pkrtz(u0[i], u1[i]);
        *(v2r*)&sVt[(dstart + i) * KB + 2 * kp] = pv;
      }

      if (kb + 1 < S_ / KB) {  // hide next tile's HBM latency behind compute
        __builtin_prefetch(g0 + KB * D_, 0, 0);
        __builtin_prefetch(h0 + KB * D_, 0, 0);
      }
    }
    __syncthreads();

    // ---- scores: S2 = (Q*log2e/8) * K^T, A-major order with B preload ----
    v8f s[4] = { vzero, vzero, vzero, vzero };
#pragma unroll
    for (int c = 0; c < 2; ++c) {
      v16h bk[4];
#pragma unroll
      for (int nb = 0; nb < 4; ++nb)
        bk[nb] = *(const v16h*)&sK[(nb * 16 + col) * D_ + c * 32 + half * 16];
#pragma unroll
      for (int nb = 0; nb < 4; ++nb)
        s[nb] = __builtin_amdgcn_wmma_f32_16x16x32_f16(false, aQ[c].v, false, bk[nb],
                                                       (short)0, s[nb], false, false);
    }

    // ---- mask (C-layout: VGPR r -> row r + 8*half, lane -> key col) ----
    float f[4][8];
#pragma unroll
    for (int nb = 0; nb < 4; ++nb)
#pragma unroll
      for (int r = 0; r < 8; ++r) {
        const int qrow = qBase + r + 8 * half;
        const int mval = M[((size_t)b * S_ + qrow) * (size_t)S_ + kBase + nb * 16 + col];
        f[nb][r] = (mval != 0) ? s[nb][r] : NEGM2;
      }

    // ---- row max: packed-f16 cross-lane reduction (2 rows per register) ----
    float bm[8], rowmax[8];
#pragma unroll
    for (int r = 0; r < 8; ++r)
      bm[r] = fmaxf(fmaxf(f[0][r], f[1][r]), fmaxf(f[2][r], f[3][r]));
#pragma unroll
    for (int rp = 0; rp < 4; ++rp) {
      HalfU u;
      u.raw = __builtin_amdgcn_cvt_pkrtz(bm[2*rp], bm[2*rp+1]);
#pragma unroll
      for (int off = 1; off < 16; off <<= 1) {   // stays within the 16-lane half
        HalfU t; t.i = __shfl_xor(u.i, off);
        u.v = __builtin_elementwise_max(u.v, t.v);   // v_pk_max_num_f16
      }
      rowmax[2*rp]   = (float)u.v[0];
      rowmax[2*rp+1] = (float)u.v[1];
    }

    // ---- online softmax: exp2, P -> LDS, rescale O ----
    float corr[8];
#pragma unroll
    for (int r = 0; r < 8; ++r) {
      const float mn = fmaxf(mrow[r], rowmax[r]);
      corr[r] = __builtin_amdgcn_exp2f(mrow[r] - mn);  // exp2(-inf)=0 on first tile
      mrow[r] = mn;

      const int prow = r + 8 * half;
#pragma unroll
      for (int nb = 0; nb < 4; ++nb) {
        const float p = __builtin_amdgcn_exp2f(f[nb][r] - mn);
        sP[wave][prow * KB + nb * 16 + col] = (_Float16)p;
      }
#pragma unroll
      for (int c = 0; c < 4; ++c) acco[c][r] *= corr[r];
    }
    // same-wave LDS store->load is in-order; compiler tracks DScnt

    // ---- P A-operand fragments (2 frags cover 64 keys) ----
    AFrag aP[2];
    {
      const int kb8 = half * 16;
#pragma unroll
      for (int j = 0; j < 2; ++j) {
        aP[j].h[0] = *(const v8h*)&sP[wave][col * KB + j * 32 + kb8];
        aP[j].h[1] = *(const v8h*)&sP[wave][col * KB + j * 32 + kb8 + 8];
      }
    }

    // ---- row sums of P via WMMA against all-ones B (no lane shuffles) ----
    v8f lacc = __builtin_amdgcn_wmma_f32_16x16x32_f16(false, aP[0].v, false, bones,
                                                      (short)0, vzero, false, false);
    lacc = __builtin_amdgcn_wmma_f32_16x16x32_f16(false, aP[1].v, false, bones,
                                                  (short)0, lacc, false, false);

    // ---- O += P * V, A-major order with B preload ----
#pragma unroll
    for (int j = 0; j < 2; ++j) {
      v16h bv[4];
#pragma unroll
      for (int c = 0; c < 4; ++c)
        bv[c] = *(const v16h*)&sVt[(c * 16 + col) * KB + j * 32 + half * 16];
#pragma unroll
      for (int c = 0; c < 4; ++c)
        acco[c] = __builtin_amdgcn_wmma_f32_16x16x32_f16(false, aP[j].v, false, bv[c],
                                                         (short)0, acco[c], false, false);
    }

#pragma unroll
    for (int r = 0; r < 8; ++r) lrow[r] = lrow[r] * corr[r] + lacc[r];

    __syncthreads();   // protect sK/sVt before next fill
  }

  // ---- epilogue: O / l, store fp32 ----
#pragma unroll
  for (int c = 0; c < 4; ++c)
#pragma unroll
    for (int r = 0; r < 8; ++r) {
      const int qrow = r + 8 * half;
      const size_t off = ((size_t)bh * S_ + qBase + qrow) * D_ + c * 16 + col;
      O[off] = acco[c][r] / lrow[r];
    }
}

extern "C" void kernel_launch(void* const* d_in, const int* in_sizes, int n_in,
                              void* d_out, int out_size, void* d_ws, size_t ws_size,
                              hipStream_t stream) {
  (void)in_sizes; (void)n_in; (void)d_ws; (void)ws_size; (void)out_size;
  const float* q = (const float*)d_in[0];
  const float* k = (const float*)d_in[1];
  const float* v = (const float*)d_in[2];
  const int*   m = (const int*)d_in[3];
  float* o = (float*)d_out;

  dim3 grid(S_ / QB, B_ * H_);   // (32, 64)
  dim3 block(128);               // 4 wave32 waves
  flash_attn_f16wmma_kernel<<<grid, block, 0, stream>>>(q, k, v, m, o);
}
